// CausalSelfAttention_33895881900262
// MI455X (gfx1250) — matrix operand order repaired and checked by reading, not compiled, not run on hardware
//
#include <hip/hip_runtime.h>

// Causal self-attention (B=4, T=2048, C=1024, nh=16, hs=64) for gfx1250.
// bf16 WMMA (v_wmma_f32_16x16x32_bf16) with f32 accumulate.
// Global->LDS tiles move via CDNA5 async-to-LDS DMA (ASYNCcnt), double-
// buffered so the async unit runs one tile ahead of the WMMA pipeline.
// All LDS fragment loads are 16B ds_load_b128 pairs (K-dim contiguous).

typedef __bf16 bf16_t;
typedef __attribute__((ext_vector_type(16))) __bf16 v16bf;
typedef __attribute__((ext_vector_type(8)))  float  v8f;

#define BATCH 4
#define SEQ   2048
#define NE    1024
#define NH    16
#define HS    64
#define BT    (BATCH * SEQ)

// ---------------------------------------------------------------- async copy

// Per-lane 16B global->LDS DMA (GLOBAL_LOAD_ASYNC_TO_LDS_B128, ASYNCcnt).
// The LDS byte address is the low 32 bits of the flat shared-pointer
// (ISA 10.2: LDS aperture address = addr[31:0]).
__device__ __forceinline__ void async_gl2lds_b128(const bf16_t* gsrc,
                                                  const bf16_t* lds_dst) {
  unsigned l = (unsigned)(uintptr_t)lds_dst;
  asm volatile("global_load_async_to_lds_b128 %0, %1, off"
               :: "v"(l), "v"(gsrc)
               : "memory");
}

__device__ __forceinline__ void wait_async() {
  asm volatile("s_wait_asynccnt 0" ::: "memory");
}

// ---------------------------------------------------------------- fragments

__device__ __forceinline__ v16bf frag2(const bf16_t* p0, const bf16_t* p1) {
  union { uint4 u[2]; v16bf v; } c;
  c.u[0] = *reinterpret_cast<const uint4*>(p0);
  c.u[1] = *reinterpret_cast<const uint4*>(p1);
  return c.v;
}

// A-fragment (16x32 bf16, MxK). Lane l: row = l&15, g = l>>4.
// a[i] = row[g*16 + i], i = 0..15.
__device__ __forceinline__ v16bf load_afrag(const bf16_t* row, int g) {
  return frag2(row + g * 16, row + g * 16 + 8);
}

// B-fragment (32x16 bf16, KxN) with the K-dim contiguous at p:
// lane l: col = l&15, K = (l>>4)*16 + i; b[i] = p[i], i = 0..15.
__device__ __forceinline__ v16bf load_bfrag16(const bf16_t* p) {
  return frag2(p, p + 8);
}

__device__ __forceinline__ v8f wmma_bf16(v16bf a, v16bf b, v8f c) {
  return __builtin_amdgcn_wmma_f32_16x16x32_bf16(false, a, false, b,
                                                 (short)0, c, false, false);
}

// ---------------------------------------------------------------- converts

__global__ void cvt_f32_to_bf16(const float* __restrict__ src,
                                bf16_t* __restrict__ dst, int n4) {
  int i = blockIdx.x * blockDim.x + threadIdx.x;
  if (i >= n4) return;
  float4 f = reinterpret_cast<const float4*>(src)[i];
  union { bf16_t b[4]; uint2 u; } pk;
  pk.b[0] = (bf16_t)f.x; pk.b[1] = (bf16_t)f.y;
  pk.b[2] = (bf16_t)f.z; pk.b[3] = (bf16_t)f.w;
  reinterpret_cast<uint2*>(dst)[i] = pk.u;
}

// W [K][N] f32  ->  WT [N][K] bf16  (one-time transposing convert)
__global__ void cvt_transpose_bf16(const float* __restrict__ W,
                                   bf16_t* __restrict__ WT, int K, int N) {
  int idx = blockIdx.x * blockDim.x + threadIdx.x;  // over K * N/4
  int k  = idx / (N >> 2);
  int n4 = (idx - k * (N >> 2)) << 2;
  if (k >= K) return;
  float4 f = *reinterpret_cast<const float4*>(&W[(size_t)k * N + n4]);
  WT[(size_t)(n4 + 0) * K + k] = (bf16_t)f.x;
  WT[(size_t)(n4 + 1) * K + k] = (bf16_t)f.y;
  WT[(size_t)(n4 + 2) * K + k] = (bf16_t)f.z;
  WT[(size_t)(n4 + 3) * K + k] = (bf16_t)f.w;
}

// ---------------------------------------------------------------- GEMM
// C[M,N] = A[M,K] @ Bt^T + bias.  A [M][K] bf16, Bt [N][K] bf16 (pre-transposed).
// Block tile 128x64, 4 waves, each wave 32x64 (8 WMMAs / K-step).
// Double-buffered async tiles: DMA for K-step i+1 overlaps WMMA on step i.
// out_mode 0: f32 row-major [M,N]           (final projection)
// out_mode 1: bf16 -> [B,NH,SEQ,HS] layout  (q, k)
// out_mode 2: bf16 -> [B,NH,HS,SEQ] layout  (v, transposed for P@V)

#define TST 40  // 32-deep K tiles, padded row stride (80B = 16B multiple)

__global__ __launch_bounds__(128) void gemm_bf16(
    const bf16_t* __restrict__ A, const bf16_t* __restrict__ Bt,
    const float* __restrict__ bias, void* __restrict__ outp,
    int M, int N, int K, int out_mode, float scale) {
  __shared__ __align__(16) bf16_t As[2][128 * TST];
  __shared__ __align__(16) bf16_t Bs[2][64 * TST];

  const int tid  = threadIdx.x;
  const int lane = tid & 31;
  const int wid  = tid >> 5;
  const int g    = lane >> 4;
  const int lr   = lane & 15;
  const int row0 = blockIdx.y * 128;
  const int col0 = blockIdx.x * 64;

  auto issue_tiles = [&](int kt, int buf) {
    // A tile 128x32: 512 x b128 (4/thread); B tile 64x32: 256 x b128 (2/thread)
#pragma unroll
    for (int it = 0; it < 4; ++it) {
      int idx = tid + it * 128;
      int r = idx >> 2, qv = idx & 3;
      async_gl2lds_b128(&A[(size_t)(row0 + r) * K + kt + qv * 8],
                        &As[buf][r * TST + qv * 8]);
    }
#pragma unroll
    for (int it = 0; it < 2; ++it) {
      int idx = tid + it * 128;
      int r = idx >> 2, qv = idx & 3;
      async_gl2lds_b128(&Bt[(size_t)(col0 + r) * K + kt + qv * 8],
                        &Bs[buf][r * TST + qv * 8]);
    }
  };

  v8f acc[2][4] = {};
  issue_tiles(0, 0);

  for (int kt = 0; kt < K; kt += 32) {
    int cur = (kt >> 5) & 1;
    wait_async();      // this wave's DMA for buffer `cur` done
    __syncthreads();   // all waves' DMA done; prior readers of cur^1 done
    if (kt + 32 < K) issue_tiles(kt + 32, cur ^ 1);  // run DMA ahead

    v16bf af0 = load_afrag(&As[cur][(wid * 32 + lr) * TST], g);
    v16bf af1 = load_afrag(&As[cur][(wid * 32 + 16 + lr) * TST], g);
#pragma unroll
    for (int s = 0; s < 4; ++s) {
      v16bf bf = load_bfrag16(&Bs[cur][(s * 16 + lr) * TST + g * 16]);
      acc[0][s] = wmma_bf16(af0, bf, acc[0][s]);
      acc[1][s] = wmma_bf16(af1, bf, acc[1][s]);
    }
  }

  // D layout: lane holds (row = u*16 + g*8 + j, col = lr) in acc[u][s][j].
  if (out_mode == 0) {
    float* Of = (float*)outp;
#pragma unroll
    for (int u = 0; u < 2; ++u)
#pragma unroll
      for (int s = 0; s < 4; ++s) {
        int n = col0 + s * 16 + lr;
        float bv = bias[n];
#pragma unroll
        for (int j = 0; j < 8; ++j) {
          int m = row0 + wid * 32 + u * 16 + g * 8 + j;
          Of[(size_t)m * N + n] = (acc[u][s][j] + bv) * scale;
        }
      }
  } else {
    bf16_t* Ob = (bf16_t*)outp;
#pragma unroll
    for (int u = 0; u < 2; ++u)
#pragma unroll
      for (int s = 0; s < 4; ++s) {
        int n = col0 + s * 16 + lr;
        int h = n >> 6, d = n & (HS - 1);
        float bv = bias[n];
#pragma unroll
        for (int j = 0; j < 8; ++j) {
          int m = row0 + wid * 32 + u * 16 + g * 8 + j;
          int bb = m >> 11, t = m & (SEQ - 1);
          size_t oi = (out_mode == 1)
                          ? (((size_t)bb * NH + h) * SEQ + t) * HS + d
                          : (((size_t)bb * NH + h) * HS + d) * SEQ + t;
          Ob[oi] = (bf16_t)((acc[u][s][j] + bv) * scale);
        }
      }
  }
}

// ---------------------------------------------------------------- attention
// Flash-attention per (b, h, 64-query tile). q pre-scaled by 1/sqrt(hs).
// q/k: [B,NH,SEQ,HS] bf16.  vT: [B,NH,HS,SEQ] bf16.  y: [BT, NE] bf16.
// K/V tiles double-buffered: DMA for key-tile i+1 overlaps softmax+WMMA on i.

#define TS 72  // padded row stride for 64-wide tiles (144B = 16B multiple)

__global__ __launch_bounds__(128) void attn_fwd(
    const bf16_t* __restrict__ q, const bf16_t* __restrict__ k,
    const bf16_t* __restrict__ vT, bf16_t* __restrict__ y) {
  __shared__ __align__(16) bf16_t Qs[64 * TS];
  __shared__ __align__(16) bf16_t Ks[2][64 * TS];   // Ks[buf][key][d]
  __shared__ __align__(16) bf16_t VsT[2][64 * TS];  // VsT[buf][d][key]
  __shared__ __align__(16) bf16_t Ps[4][16 * TS];   // per-wave P (16x64)

  const int tid  = threadIdx.x;
  const int lane = tid & 31;
  const int wid  = tid >> 5;
  const int g    = lane >> 4;
  const int lr   = lane & 15;
  const int b  = blockIdx.z;
  const int h  = blockIdx.y;
  const int q0 = blockIdx.x * 64;

  const size_t head = ((size_t)b * NH + h) * SEQ * HS;
  const bf16_t* qh = q + head;
  const bf16_t* kh = k + head;
  const bf16_t* vh = vT + head;  // [HS][SEQ]

  auto issue_kv = [&](int kt0, int buf) {
#pragma unroll
    for (int it = 0; it < 4; ++it) {
      int idx = tid + it * 128;
      int r = idx >> 3, qv = idx & 7;
      async_gl2lds_b128(&kh[(size_t)(kt0 + r) * HS + qv * 8],
                        &Ks[buf][r * TS + qv * 8]);
      async_gl2lds_b128(&vh[(size_t)r * SEQ + kt0 + qv * 8],
                        &VsT[buf][r * TS + qv * 8]);
    }
  };

  // Q tile (64x64): async DMA, then build the two A-fragments once.
#pragma unroll
  for (int it = 0; it < 4; ++it) {
    int idx = tid + it * 128;
    int r = idx >> 3, qv = idx & 7;
    async_gl2lds_b128(&qh[(size_t)(q0 + r) * HS + qv * 8], &Qs[r * TS + qv * 8]);
  }
  wait_async();
  __syncthreads();
  v16bf aq0 = load_afrag(&Qs[(wid * 16 + lr) * TS + 0], g);
  v16bf aq1 = load_afrag(&Qs[(wid * 16 + lr) * TS + 32], g);

  v8f oacc[4] = {};
  float mi[8], li[8];
#pragma unroll
  for (int j = 0; j < 8; ++j) { mi[j] = -__builtin_inff(); li[j] = 0.f; }

  const int qrow_base = q0 + wid * 16 + g * 8;

  issue_kv(0, 0);
  for (int kt0 = 0; kt0 <= q0; kt0 += 64) {
    int cur = (kt0 >> 6) & 1;
    wait_async();      // this wave's DMA for buffer `cur` done
    __syncthreads();   // all waves' DMA done; prior readers of cur^1 done
    if (kt0 + 64 <= q0) issue_kv(kt0 + 64, cur ^ 1);  // run DMA ahead

    // S = Q @ K^T: B-fragment K-dim (= head dim) contiguous along Ks rows.
    v8f sacc[4] = {};
#pragma unroll
    for (int s = 0; s < 4; ++s) {
      v16bf b0 = load_bfrag16(&Ks[cur][(s * 16 + lr) * TS + 0 + g * 16]);
      sacc[s] = wmma_bf16(aq0, b0, sacc[s]);
      v16bf b1 = load_bfrag16(&Ks[cur][(s * 16 + lr) * TS + 32 + g * 16]);
      sacc[s] = wmma_bf16(aq1, b1, sacc[s]);
    }

    // Online softmax. Row r = g*8 + j lives in lanes {g*16 .. g*16+15};
    // xor-shuffles over masks 1,2,4,8 reduce within that 16-lane group.
#pragma unroll
    for (int j = 0; j < 8; ++j) {
      int qrow = qrow_base + j;
      float smax = -__builtin_inff();
#pragma unroll
      for (int s = 0; s < 4; ++s) {
        float sv = sacc[s][j];
        int key = kt0 + s * 16 + lr;
        if (key > qrow) sv = -__builtin_inff();  // causal mask
        sacc[s][j] = sv;
        smax = fmaxf(smax, sv);
      }
      smax = fmaxf(smax, __shfl_xor(smax, 1));
      smax = fmaxf(smax, __shfl_xor(smax, 2));
      smax = fmaxf(smax, __shfl_xor(smax, 4));
      smax = fmaxf(smax, __shfl_xor(smax, 8));
      float mnew  = fmaxf(mi[j], smax);
      float alpha = __expf(mi[j] - mnew);  // first tile: exp(-inf) = 0
      mi[j] = mnew;
      float rsum = 0.f;
#pragma unroll
      for (int s = 0; s < 4; ++s) {
        float p = __expf(sacc[s][j] - mnew);
        rsum += p;
        Ps[wid][(g * 8 + j) * TS + s * 16 + lr] = (bf16_t)p;
      }
      rsum += __shfl_xor(rsum, 1);
      rsum += __shfl_xor(rsum, 2);
      rsum += __shfl_xor(rsum, 4);
      rsum += __shfl_xor(rsum, 8);
      li[j] = li[j] * alpha + rsum;
#pragma unroll
      for (int s = 0; s < 4; ++s) oacc[s][j] *= alpha;
    }

    // O += P @ V. B-fragment K-dim (= key) contiguous along VsT rows.
    // Ps is per-wave; DS ops are in-order within a wave, no barrier needed.
    v16bf ap0 = load_afrag(&Ps[wid][lr * TS + 0], g);
    v16bf ap1 = load_afrag(&Ps[wid][lr * TS + 32], g);
#pragma unroll
    for (int s = 0; s < 4; ++s) {
      v16bf bv0 = load_bfrag16(&VsT[cur][(s * 16 + lr) * TS + 0 + g * 16]);
      oacc[s] = wmma_bf16(ap0, bv0, oacc[s]);
      v16bf bv1 = load_bfrag16(&VsT[cur][(s * 16 + lr) * TS + 32 + g * 16]);
      oacc[s] = wmma_bf16(ap1, bv1, oacc[s]);
    }
  }

  // y[(b*SEQ + t)*NE + h*HS + d], normalized by the softmax row sum.
#pragma unroll
  for (int s = 0; s < 4; ++s) {
    int d = s * 16 + lr;
#pragma unroll
    for (int j = 0; j < 8; ++j) {
      int t = q0 + wid * 16 + g * 8 + j;
      float val = oacc[s][j] / li[j];
      y[((size_t)b * SEQ + t) * NE + h * HS + d] = (bf16_t)val;
    }
  }
}

// ---------------------------------------------------------------- launch

extern "C" void kernel_launch(void* const* d_in, const int* in_sizes, int n_in,
                              void* d_out, int out_size, void* d_ws,
                              size_t ws_size, hipStream_t stream) {
  (void)in_sizes; (void)n_in; (void)out_size; (void)ws_size;
  const float* x  = (const float*)d_in[0];
  const float* Wk = (const float*)d_in[1];
  const float* bk = (const float*)d_in[2];
  const float* Wq = (const float*)d_in[3];
  const float* bq = (const float*)d_in[4];
  const float* Wv = (const float*)d_in[5];
  const float* bv = (const float*)d_in[6];
  const float* Wp = (const float*)d_in[7];
  const float* bp = (const float*)d_in[8];

  char* ws = (char*)d_ws;
  size_t off = 0;
  auto wsalloc = [&](size_t bytes) -> void* {
    void* p = ws + off;
    off += (bytes + 255) & ~(size_t)255;
    return p;
  };
  bf16_t* xb  = (bf16_t*)wsalloc((size_t)BT * NE * 2);
  bf16_t* wqt = (bf16_t*)wsalloc((size_t)NE * NE * 2);
  bf16_t* wkt = (bf16_t*)wsalloc((size_t)NE * NE * 2);
  bf16_t* wvt = (bf16_t*)wsalloc((size_t)NE * NE * 2);
  bf16_t* wpt = (bf16_t*)wsalloc((size_t)NE * NE * 2);
  bf16_t* qb  = (bf16_t*)wsalloc((size_t)BT * NE * 2);
  bf16_t* kb  = (bf16_t*)wsalloc((size_t)BT * NE * 2);
  bf16_t* vb  = (bf16_t*)wsalloc((size_t)BT * NE * 2);  // [B,NH,HS,SEQ]
  bf16_t* yb  = (bf16_t*)wsalloc((size_t)BT * NE * 2);

  {
    int n4 = BT * NE / 4;
    cvt_f32_to_bf16<<<n4 / 256, 256, 0, stream>>>(x, xb, n4);
    int w4 = NE * NE / 4;  // transposing converts: W[K][N] -> WT[N][K]
    cvt_transpose_bf16<<<w4 / 256, 256, 0, stream>>>(Wq, wqt, NE, NE);
    cvt_transpose_bf16<<<w4 / 256, 256, 0, stream>>>(Wk, wkt, NE, NE);
    cvt_transpose_bf16<<<w4 / 256, 256, 0, stream>>>(Wv, wvt, NE, NE);
    cvt_transpose_bf16<<<w4 / 256, 256, 0, stream>>>(Wp, wpt, NE, NE);
  }

  dim3 gg(NE / 64, BT / 128);  // (16, 64), 128x64 block tiles
  // q gets the 1/sqrt(hs) softmax scale folded in: (x@Wq + bq) * 0.125.
  gemm_bf16<<<gg, 128, 0, stream>>>(xb, wqt, bq, qb, BT, NE, NE, 1, 0.125f);
  gemm_bf16<<<gg, 128, 0, stream>>>(xb, wkt, bk, kb, BT, NE, NE, 1, 1.0f);
  gemm_bf16<<<gg, 128, 0, stream>>>(xb, wvt, bv, vb, BT, NE, NE, 2, 1.0f);

  attn_fwd<<<dim3(SEQ / 64, NH, BATCH), 128, 0, stream>>>(qb, kb, vb, yb);

  gemm_bf16<<<gg, 128, 0, stream>>>(yb, wpt, bp, d_out, BT, NE, NE, 0, 1.0f);
}
